// GraphAttentionLayer_9457517985826
// MI455X (gfx1250) — matrix operand order repaired and checked by reading, not compiled, not run on hardware
//
#include <hip/hip_runtime.h>

// ---------------------------------------------------------------------------
// GAT layer, fused flash-attention style, bf16 WMMA on gfx1250 (wave32)
//   B=4, N=4096, Fin=256, Fout=64
// adj (268 MB, read-once) streamed with non-temporal hints so L2 stays
// dedicated to WhT / Wh2 which are re-read 256x per batch.
// ---------------------------------------------------------------------------

typedef __attribute__((ext_vector_type(16))) __bf16 v16bf;
typedef __attribute__((ext_vector_type(8)))  __bf16 v8bf;
typedef __attribute__((ext_vector_type(4)))  __bf16 v4bf;
typedef __attribute__((ext_vector_type(8)))  float  v8f;
typedef __attribute__((ext_vector_type(4)))  float  v4f;
typedef __attribute__((ext_vector_type(4)))  int    v4i;

constexpr int   B_   = 4;
constexpr int   N_   = 4096;
constexpr int   FIN  = 256;
constexpr int   FOUT = 64;
constexpr float ALPHA   = 0.2f;
constexpr float NEG_BIG = -9e15f;

// ---- kernel 0a: h (f32) -> hbf (bf16), 4 elems/thread -----------------------
__global__ void k_conv_h(const float* __restrict__ h, __bf16* __restrict__ hbf, int n4) {
    int i = blockIdx.x * blockDim.x + threadIdx.x;
    if (i >= n4) return;
    v4f x = __builtin_nontemporal_load((const v4f*)(h + 4 * (size_t)i)); // read-once
    v4bf y;
#pragma unroll
    for (int k = 0; k < 4; ++k) y[k] = (__bf16)x[k];
    *(v4bf*)(hbf + 4 * (size_t)i) = y;   // re-read by GEMM -> keep cached
}

// ---- kernel 0b: W [256,64] f32 -> WT [64,256] bf16 (transpose) --------------
__global__ void k_conv_w(const float* __restrict__ W, __bf16* __restrict__ WT) {
    int i = blockIdx.x * blockDim.x + threadIdx.x;   // 16384 threads
    int n = i & (FOUT - 1);
    int k = i >> 6;
    WT[(size_t)n * FIN + k] = (__bf16)W[i];          // coalesced read, scatter write
}

// ---- kernel 1: Wh = h @ W  via bf16 WMMA, stored transposed WhT[b][f][row] --
// one wave per 16-row x 64-col output tile; 8 K-steps of 32
__global__ void __launch_bounds__(128) k_gemm_wh(const __bf16* __restrict__ hbf,
                                                 const __bf16* __restrict__ WT,
                                                 __bf16* __restrict__ WhT) {
    const int lane = threadIdx.x & 31;
    const int wave = threadIdx.x >> 5;
    const int tile = blockIdx.x * 4 + wave;
    const int b    = tile >> 8;              // 256 row-tiles per batch
    const int i0   = (tile & 255) << 4;
    const int g    = lane >> 4;
    const int mn   = lane & 15;

    v8f acc[4] = {};
    const __bf16* hrow = hbf + ((size_t)b * N_ + i0 + mn) * FIN;

#pragma unroll
    for (int kk = 0; kk < 8; ++kk) {
        const int kbase = kk * 32;
        // A-fragment: lane (g,m) holds K run [kbase+16g, +16)
        v8bf a0 = *(const v8bf*)(hrow + kbase + 16 * g);
        v8bf a1 = *(const v8bf*)(hrow + kbase + 16 * g + 8);
        v16bf A;
#pragma unroll
        for (int e = 0; e < 8; ++e) { A[e] = a0[e]; A[8 + e] = a1[e]; }
#pragma unroll
        for (int t = 0; t < 4; ++t) {
            const int col = 16 * t + mn;
            // B-fragment: lane (g,n) holds W[K=kbase+16g .. +16)[col] = WT row slice
            v16bf Bf = *(const v16bf*)(WT + (size_t)col * FIN + kbase + 16 * g);
            acc[t] = __builtin_amdgcn_wmma_f32_16x16x32_bf16(
                false, A, false, Bf, (short)0, acc[t], false, false);
        }
    }
    // D layout: lane (g,n), vgpr v -> row i0+v+8g, col 16t+n. Store transposed.
#pragma unroll
    for (int t = 0; t < 4; ++t) {
        v8bf o;
#pragma unroll
        for (int v = 0; v < 8; ++v) o[v] = (__bf16)acc[t][v];
        *(v8bf*)(WhT + ((size_t)b * FOUT + 16 * t + mn) * N_ + i0 + 8 * g) = o;
    }
}

// ---- kernel 2: Wh1 = Wh . a[:64], Wh2 = Wh . a[64:] -------------------------
__global__ void k_rowdot(const __bf16* __restrict__ WhT, const float* __restrict__ a,
                         float* __restrict__ Wh1, float* __restrict__ Wh2) {
    int i = blockIdx.x * blockDim.x + threadIdx.x;   // B*N threads
    int b = i >> 12;
    int r = i & (N_ - 1);
    const __bf16* base = WhT + (size_t)b * FOUT * N_ + r;   // coalesced over r
    float s1 = 0.f, s2 = 0.f;
#pragma unroll 8
    for (int f = 0; f < FOUT; ++f) {
        float v = (float)base[(size_t)f * N_];
        s1 += v * a[f];
        s2 += v * a[FOUT + f];
    }
    Wh1[i] = s1;
    Wh2[i] = s2;
}

// ---- kernel 3: fused masked-softmax(e) @ Wh with online (flash) softmax -----
// one wave per 16-row tile; j swept in steps of 32; adj streamed once (268 MB)
__global__ void __launch_bounds__(128) k_attn(const int* __restrict__ adj,
                                              const __bf16* __restrict__ WhT,
                                              const float* __restrict__ Wh1,
                                              const float* __restrict__ Wh2,
                                              float* __restrict__ out) {
    const int lane = threadIdx.x & 31;
    const int wave = threadIdx.x >> 5;
    const int tile = blockIdx.x * 4 + wave;
    const int b    = tile >> 8;
    const int i0   = (tile & 255) << 4;
    const int g    = lane >> 4;
    const int mn   = lane & 15;     // A-layout row / B,D-layout column

    const float  s1     = Wh1[(size_t)b * N_ + i0 + mn];
    const float* wh2    = Wh2 + (size_t)b * N_;
    const int*   adjrow = adj + ((size_t)b * N_ + i0 + mn) * (size_t)N_;
    const __bf16* whtb  = WhT + (size_t)b * FOUT * N_;

    float mrun = NEG_BIG, lrun = 0.f;
    v8f acc[4] = {};

    for (int j0 = 0; j0 < N_; j0 += 32) {
        const int jA = j0 + 16 * g;         // K run 0 of A-fragment
        const int jB = j0 + 16 * g + 8;     // K run 1 of A-fragment
        v4f wA0 = *(const v4f*)(wh2 + jA);
        v4f wA1 = *(const v4f*)(wh2 + jA + 4);
        v4f wB0 = *(const v4f*)(wh2 + jB);
        v4f wB1 = *(const v4f*)(wh2 + jB + 4);
        // adj: one-shot 268 MB stream -> non-temporal (don't thrash L2)
        v4i aA0 = __builtin_nontemporal_load((const v4i*)(adjrow + jA));
        v4i aA1 = __builtin_nontemporal_load((const v4i*)(adjrow + jA + 4));
        v4i aB0 = __builtin_nontemporal_load((const v4i*)(adjrow + jB));
        v4i aB1 = __builtin_nontemporal_load((const v4i*)(adjrow + jB + 4));
        if (j0 + 32 < N_) {                 // global_prefetch_b8 next adj tile
            __builtin_prefetch(adjrow + jA + 32, 0, 0);
            __builtin_prefetch(adjrow + jB + 32, 0, 0);
        }

        // scores: leaky_relu(Wh1[i]+Wh2[j]) masked -> NEG_BIG (matches torch fill)
        float e[16];
        float lm = NEG_BIG;
#pragma unroll
        for (int k = 0; k < 8; ++k) {
            float w  = (k < 4) ? wA0[k] : wA1[k - 4];
            int   ad = (k < 4) ? aA0[k] : aA1[k - 4];
            float s  = s1 + w;
            s = (s >= 0.f) ? s : ALPHA * s;
            e[k] = (ad > 0) ? s : NEG_BIG;
            lm = fmaxf(lm, e[k]);
        }
#pragma unroll
        for (int k = 0; k < 8; ++k) {
            float w  = (k < 4) ? wB0[k] : wB1[k - 4];
            int   ad = (k < 4) ? aB0[k] : aB1[k - 4];
            float s  = s1 + w;
            s = (s >= 0.f) ? s : ALPHA * s;
            e[8 + k] = (ad > 0) ? s : NEG_BIG;
            lm = fmaxf(lm, e[8 + k]);
        }
        // row max/sum live in lane pair (m, m+16)
        lm = fmaxf(lm, __shfl_xor(lm, 16));
        const float newm = fmaxf(mrun, lm);
        float p[16], psum = 0.f;
#pragma unroll
        for (int k = 0; k < 16; ++k) { p[k] = __expf(e[k] - newm); psum += p[k]; }
        psum += __shfl_xor(psum, 16);
        const float corr = __expf(mrun - newm);
        lrun = lrun * corr + psum;
        mrun = newm;

        // rescale accumulator: D-layout row of vgpr v is (v + 8g); corr for row r
        // lives in lane r -> broadcast via ds_bpermute
        float sc[8];
#pragma unroll
        for (int v = 0; v < 8; ++v) sc[v] = __shfl(corr, 8 * g + v);
#pragma unroll
        for (int t = 0; t < 4; ++t)
#pragma unroll
            for (int v = 0; v < 8; ++v) acc[t][v] *= sc[v];

        // P (16x32) -> bf16 A-fragment (lane already holds its 16 K slots in order)
        v16bf P;
#pragma unroll
        for (int k = 0; k < 16; ++k) P[k] = (__bf16)p[k];

#pragma unroll
        for (int t = 0; t < 4; ++t) {
            const int col = 16 * t + mn;
            v16bf Bf = *(const v16bf*)(whtb + (size_t)col * N_ + j0 + 16 * g);
            acc[t] = __builtin_amdgcn_wmma_f32_16x16x32_bf16(
                false, P, false, Bf, (short)0, acc[t], false, false);
        }
    }

    // finalize: divide by row sum, ELU, store f32 output [B,N,64] (write-once -> NT)
    const float rinv = 1.0f / lrun;
    float rv[8];
#pragma unroll
    for (int v = 0; v < 8; ++v) rv[v] = __shfl(rinv, 8 * g + v);
#pragma unroll
    for (int t = 0; t < 4; ++t) {
#pragma unroll
        for (int v = 0; v < 8; ++v) {
            float o = acc[t][v] * rv[v];
            o = (o > 0.f) ? o : (__expf(o) - 1.f);
            __builtin_nontemporal_store(
                o, out + ((size_t)b * N_ + i0 + 8 * g + v) * FOUT + 16 * t + mn);
        }
    }
}

// ---------------------------------------------------------------------------
extern "C" void kernel_launch(void* const* d_in, const int* in_sizes, int n_in,
                              void* d_out, int out_size, void* d_ws, size_t ws_size,
                              hipStream_t stream) {
    const float* h   = (const float*)d_in[0];
    const int*   adj = (const int*)d_in[1];
    const float* W   = (const float*)d_in[2];
    const float* a   = (const float*)d_in[3];
    float*       out = (float*)d_out;

    // workspace layout (all offsets 256B-aligned): ~10.6 MB total
    char* ws = (char*)d_ws;
    size_t off = 0;
    __bf16* hbf = (__bf16*)(ws + off); off += (size_t)B_ * N_ * FIN * 2;   // 8 MB
    __bf16* WT  = (__bf16*)(ws + off); off += (size_t)FOUT * FIN * 2;      // 32 KB
    __bf16* WhT = (__bf16*)(ws + off); off += (size_t)B_ * FOUT * N_ * 2;  // 2 MB
    float*  Wh1 = (float*)(ws + off);  off += (size_t)B_ * N_ * 4;         // 64 KB
    float*  Wh2 = (float*)(ws + off);  off += (size_t)B_ * N_ * 4;         // 64 KB
    (void)ws_size; (void)in_sizes; (void)n_in; (void)out_size;

    const int n4 = B_ * N_ * FIN / 4;
    k_conv_h <<<(n4 + 255) / 256, 256, 0, stream>>>(h, hbf, n4);
    k_conv_w <<<(FIN * FOUT) / 256, 256, 0, stream>>>(W, WT);
    k_gemm_wh<<<B_ * (N_ / 16) / 4, 128, 0, stream>>>(hbf, WT, WhT);
    k_rowdot <<<B_ * N_ / 256, 256, 0, stream>>>(WhT, a, Wh1, Wh2);
    k_attn   <<<B_ * (N_ / 16) / 4, 128, 0, stream>>>(adj, WhT, Wh1, Wh2, out);
}
